// DeepSpeedSelfAttention_49125835931688
// MI455X (gfx1250) — matrix operand order repaired and checked by reading, not compiled, not run on hardware
//
#include <hip/hip_runtime.h>

// ---------------- problem constants ----------------
constexpr int B_ = 2;
constexpr int S_ = 2048;
constexpr int H_ = 2048;
constexpr int HEADS_ = 16;
constexpr int DH_ = 128;               // H_/HEADS_
constexpr int H3_ = 3 * H_;            // 6144
constexpr int ROWS_ = B_ * S_;         // 4096
constexpr float MINUS_INF = -10000.0f;
constexpr float LN_EPS = 1e-12f;

// LDS staging geometry for the block GEMMs
constexpr int BROWS = 64;              // B^T rows (N) staged per block
constexpr int KSTG = 64;               // K depth per stage
constexpr int BSTRIDE = 72;            // padded row stride in shorts (144B, 16B aligned)

// ---------------- CDNA5 WMMA types ----------------
typedef __attribute__((ext_vector_type(16))) __bf16 v16bf;
typedef __attribute__((ext_vector_type(8)))  float  v8f;

union FragU {
  uint4 q[2];
  unsigned int u[8];
  v16bf v;
};

__device__ __forceinline__ unsigned short f2bf(float f) {
  union { float f; unsigned u; } x; x.f = f;
  unsigned r = x.u + 0x7FFFu + ((x.u >> 16) & 1u);   // round-to-nearest-even
  return (unsigned short)(r >> 16);
}

__device__ __forceinline__ v8f vzero8() {
  v8f z = {0.f, 0.f, 0.f, 0.f, 0.f, 0.f, 0.f, 0.f};
  return z;
}

// A-tile 16x32 (MxK), bf16, from row-major [16 x lda] source.
// ISA 7.12.2: lanes 0-15 -> M=lane, K 0-7 (v0-3) & 16-23 (v4-7);
//             lanes 16-31 -> M=lane-16, K 8-15 & 24-31.
__device__ __forceinline__ v16bf load_frag_a(const unsigned short* base, int lda) {
  const int lane = threadIdx.x & 31;
  const int m = lane & 15, hi = lane >> 4;
  const unsigned short* p = base + (size_t)m * lda;
  FragU f;
  f.q[0] = *(const uint4*)(p + hi * 8);
  f.q[1] = *(const uint4*)(p + 16 + hi * 8);
  return f.v;
}

// B-tile 32x16 (KxN), bf16, loaded from B^T stored row-major [N x ldb]:
// lane n owns column N=n; low half K 0-7 & 16-23, high half K 8-15 & 24-31.
__device__ __forceinline__ v16bf load_frag_bt(const unsigned short* baseT, int ldb) {
  const int lane = threadIdx.x & 31;
  const int n = lane & 15, hi = lane >> 4;
  const unsigned short* p = baseT + (size_t)n * ldb;
  FragU f;
  f.q[0] = *(const uint4*)(p + hi * 8);
  f.q[1] = *(const uint4*)(p + 16 + hi * 8);
  return f.v;
}

__device__ __forceinline__ v8f wmma_bf16(v16bf a, v16bf b, v8f c) {
  return __builtin_amdgcn_wmma_f32_16x16x32_bf16(false, a, false, b, (short)0, c,
                                                 false, false);
}

// ---------------- CDNA5 async global->LDS copy (ASYNCcnt-tracked) ----------------
__device__ __forceinline__ void async_b128(unsigned lds_byte_off,
                                           unsigned long long gaddr) {
  asm volatile("global_load_async_to_lds_b128 %0, %1, off"
               :: "v"(lds_byte_off), "v"(gaddr)
               : "memory");
}
__device__ __forceinline__ void wait_async_le2() {
  asm volatile("s_wait_asynccnt 0x2" ::: "memory");
}
__device__ __forceinline__ void wait_async_0() {
  asm volatile("s_wait_asynccnt 0x0" ::: "memory");
}

// ---------------- shared GEMM core: block = 8 waves, tile 128(M) x 64(N) ----------
// Arow: global A row base for this wave's 16 rows (row-major, lda = H_)
// BTbase: global B^T base at row n0 (row-major, ldb = H_)
// lds: staging buffer, 2 * BROWS * BSTRIDE shorts
__device__ __forceinline__ void gemm_core_128x64(const unsigned short* __restrict__ Arow,
                                                 const unsigned short* __restrict__ BTbase,
                                                 unsigned short* lds, v8f c[4]) {
  const int tid = threadIdx.x;          // 0..255
  const int row = tid >> 2;             // 0..63  (B^T row within tile)
  const int chunk = tid & 3;            // 0..3   (32B chunk within 128B row slice)
  const unsigned lds_base = (unsigned)(uintptr_t)lds;
  const unsigned stage_bytes = BROWS * BSTRIDE * 2;

  auto issue = [&](int buf, int kk) {
    const unsigned short* g = BTbase + (size_t)row * H_ + kk + chunk * 16;
    const unsigned loff = lds_base + (unsigned)buf * stage_bytes +
                          (unsigned)row * (BSTRIDE * 2) + (unsigned)chunk * 32;
    async_b128(loff, (unsigned long long)(uintptr_t)g);
    async_b128(loff + 16, (unsigned long long)(uintptr_t)(g + 8));
  };

  auto compute = [&](int buf, int kk) {
    const unsigned short* bbuf = lds + (size_t)buf * (BROWS * BSTRIDE);
    __builtin_prefetch(Arow + kk + 256, 0, 0);
#pragma unroll
    for (int ks = 0; ks < KSTG; ks += 32) {
      v16bf a = load_frag_a(Arow + kk + ks, H_);
#pragma unroll
      for (int t = 0; t < 4; ++t) {
        v16bf b = load_frag_bt(bbuf + (size_t)(16 * t) * BSTRIDE + ks, BSTRIDE);
        c[t] = wmma_bf16(a, b, c[t]);
      }
    }
  };

  issue(0, 0);
  int buf = 0;
  int kk = 0;
  // steady state: unconditionally issue next stage, drain to <=2 outstanding
  for (; kk < H_ - KSTG; kk += KSTG, buf ^= 1) {
    issue(buf ^ 1, kk + KSTG);
    wait_async_le2();
    __syncthreads();                    // stage `buf` visible to all 8 waves
    compute(buf, kk);
    __syncthreads();                    // all waves done reading `buf`
  }
  // peeled final stage
  wait_async_0();
  __syncthreads();
  compute(buf, kk);
}

// ---------------- kernel 1: fused layernorm ----------------
__global__ __launch_bounds__(256) void ln_kernel(const float* __restrict__ x,
                                                 const float* __restrict__ g,
                                                 const float* __restrict__ bta,
                                                 float* __restrict__ xn_out,
                                                 unsigned short* __restrict__ xn_bf) {
  const int r = blockIdx.x;
  const int tid = threadIdx.x;
  const float* xr = x + (size_t)r * H_;
  float s = 0.f, s2 = 0.f;
  for (int i = tid; i < H_; i += 256) {
    float v = xr[i];
    s += v; s2 += v * v;
  }
  __shared__ float red0[256];
  __shared__ float red1[256];
  red0[tid] = s; red1[tid] = s2;
  __syncthreads();
  for (int off = 128; off > 0; off >>= 1) {
    if (tid < off) { red0[tid] += red0[tid + off]; red1[tid] += red1[tid + off]; }
    __syncthreads();
  }
  const float mean = red0[0] * (1.0f / H_);
  const float var  = red1[0] * (1.0f / H_) - mean * mean;
  const float rstd = rsqrtf(var + LN_EPS);
  for (int i = tid; i < H_; i += 256) {
    float v = (xr[i] - mean) * rstd * g[i] + bta[i];
    xn_out[(size_t)r * H_ + i] = v;
    xn_bf[(size_t)r * H_ + i] = f2bf(v);
  }
}

// ---------------- kernel 2: weight convert + transpose to bf16 ----------------
__global__ __launch_bounds__(256) void wconv_kernel(const float* __restrict__ qkvw,
                                                    const float* __restrict__ ow,
                                                    unsigned short* __restrict__ qkvwT,
                                                    unsigned short* __restrict__ owT) {
  const size_t QK = (size_t)H3_ * H_;
  const size_t TOT = QK + (size_t)H_ * H_;
  const size_t j = (size_t)blockIdx.x * 256 + threadIdx.x;
  if (j >= TOT) return;
  if (j < QK) {
    const int e = (int)(j / H_);
    const int h = (int)(j % H_);
    qkvwT[j] = f2bf(qkvw[(size_t)h * H3_ + e]);
  } else {
    const size_t j2 = j - QK;
    const int o = (int)(j2 / H_);
    const int h = (int)(j2 % H_);
    owT[j2] = f2bf(ow[(size_t)h * H_ + o]);
  }
}

// ---------------- kernel 3: QKV GEMM (block: 128x64, 8 waves) ----------------
__global__ __launch_bounds__(256) void qkv_gemm_kernel(
    const unsigned short* __restrict__ xn_bf, const unsigned short* __restrict__ qkvwT,
    const float* __restrict__ qkvb,
    float* __restrict__ k_out, float* __restrict__ v_out,
    unsigned short* __restrict__ q_bf, unsigned short* __restrict__ k_bf,
    unsigned short* __restrict__ vT_bf) {
  __shared__ unsigned short bstage[2 * BROWS * BSTRIDE];
  const int wave = threadIdx.x >> 5;
  const int lane = threadIdx.x & 31;
  const int n = lane & 15, hi = lane >> 4;
  const int m0 = blockIdx.x * 128 + wave * 16;
  const int n0 = blockIdx.y * 64;

  v8f c[4];
#pragma unroll
  for (int t = 0; t < 4; ++t) c[t] = vzero8();

  gemm_core_128x64(xn_bf + (size_t)m0 * H_, qkvwT + (size_t)n0 * H_, bstage, c);

  const float qscale = 0.08838834764831845f;  // 1/sqrt(128)
#pragma unroll
  for (int t = 0; t < 4; ++t) {
    const int col = n0 + 16 * t + n;          // [0, 3H)
    const float bias = qkvb[col];
    const int which = col / H_;               // 0=q 1=k 2=v
    const int hcol = col - which * H_;
    const int headi = hcol / DH_;
    const int d = hcol % DH_;
#pragma unroll
    for (int j = 0; j < 8; ++j) {
      const int r = m0 + j + 8 * hi;
      const float val = c[t][j] + bias;
      const int b_ = r / S_, s_ = r % S_;
      const size_t kidx = (((size_t)b_ * HEADS_ + headi) * S_ + s_) * DH_ + d;
      if (which == 0) {
        q_bf[kidx] = f2bf(val * qscale);
      } else if (which == 1) {
        k_out[kidx] = val;
        k_bf[kidx] = f2bf(val);
      } else {
        v_out[kidx] = val;
        vT_bf[(((size_t)b_ * HEADS_ + headi) * DH_ + d) * S_ + s_] = f2bf(val);
      }
    }
  }
}

// ---------------- kernel 4: causal flash attention (wave: 16 queries) ----------------
__global__ __launch_bounds__(32) void attn_kernel(
    const float* __restrict__ mask, const unsigned short* __restrict__ q_bf,
    const unsigned short* __restrict__ k_bf, const unsigned short* __restrict__ vT_bf,
    float* __restrict__ ctx_out, unsigned short* __restrict__ ctx_bf) {
  const int QT = S_ / 16;
  const int bid = blockIdx.x;
  const int qt = bid % QT;
  const int head = (bid / QT) % HEADS_;
  const int b = bid / (QT * HEADS_);
  const int lane = threadIdx.x & 31;
  const int n = lane & 15, hi = lane >> 4;

  const unsigned short* qbase = q_bf + ((size_t)b * HEADS_ + head) * S_ * DH_;
  const unsigned short* kbase = k_bf + ((size_t)b * HEADS_ + head) * S_ * DH_;
  const unsigned short* vtbase = vT_bf + ((size_t)b * HEADS_ + head) * DH_ * S_;

  v16bf aq[4];
#pragma unroll
  for (int cchunk = 0; cchunk < 4; ++cchunk)
    aq[cchunk] = load_frag_a(qbase + (size_t)(qt * 16) * DH_ + cchunk * 32, DH_);

  v8f acc[8];
#pragma unroll
  for (int t = 0; t < 8; ++t) acc[t] = vzero8();
  float m_i[8], l_i[8];
#pragma unroll
  for (int j = 0; j < 8; ++j) { m_i[j] = -1e30f; l_i[j] = 0.f; }

  __shared__ unsigned short p_lds[16 * 32];

  const int klim = qt * 16 + 16;
  for (int kk = 0; kk < klim; kk += 32) {
    float s0[8], s1[8];
    {  // half 0 (always valid: kk < klim)
      v8f sc = vzero8();
      __builtin_prefetch(kbase + (size_t)(kk + 32) * DH_ + n * DH_, 0, 0);
#pragma unroll
      for (int cchunk = 0; cchunk < 4; ++cchunk)
        sc = wmma_bf16(aq[cchunk],
                       load_frag_bt(kbase + (size_t)kk * DH_ + cchunk * 32, DH_), sc);
      const float mv = mask[(size_t)b * S_ + kk + n];
#pragma unroll
      for (int j = 0; j < 8; ++j) {
        const int row = qt * 16 + j + 8 * hi;
        const int col = kk + n;
        s0[j] = (col <= row) ? (sc[j] + mv) : MINUS_INF;
      }
    }
    const int k1 = kk + 16;
    if (k1 < klim) {
      v8f sc = vzero8();
#pragma unroll
      for (int cchunk = 0; cchunk < 4; ++cchunk)
        sc = wmma_bf16(aq[cchunk],
                       load_frag_bt(kbase + (size_t)k1 * DH_ + cchunk * 32, DH_), sc);
      const float mv = mask[(size_t)b * S_ + k1 + n];
#pragma unroll
      for (int j = 0; j < 8; ++j) {
        const int row = qt * 16 + j + 8 * hi;
        const int col = k1 + n;
        s1[j] = (col <= row) ? (sc[j] + mv) : MINUS_INF;
      }
    } else {
#pragma unroll
      for (int j = 0; j < 8; ++j) s1[j] = MINUS_INF;
    }

    // online softmax: row reductions across the 16-lane half owning each row
    float corr[8];
#pragma unroll
    for (int j = 0; j < 8; ++j) {
      float r = fmaxf(s0[j], s1[j]);
#pragma unroll
      for (int t = 1; t < 16; t <<= 1) r = fmaxf(r, __shfl_xor(r, t, 32));
      const float mn = fmaxf(m_i[j], r);
      const float p0 = __expf(s0[j] - mn);
      const float p1 = __expf(s1[j] - mn);
      float sum = p0 + p1;
#pragma unroll
      for (int t = 1; t < 16; t <<= 1) sum += __shfl_xor(sum, t, 32);
      corr[j] = __expf(m_i[j] - mn);
      l_i[j] = l_i[j] * corr[j] + sum;
      m_i[j] = mn;
      const int m = j + 8 * hi;
      p_lds[m * 32 + n] = f2bf(p0);
      p_lds[m * 32 + 16 + n] = f2bf(p1);
    }
#pragma unroll
    for (int t = 0; t < 8; ++t) {
      v8f a2 = acc[t];
#pragma unroll
      for (int j = 0; j < 8; ++j) a2[j] *= corr[j];
      acc[t] = a2;
    }
    __syncthreads();
    v16bf ap = load_frag_a(p_lds, 32);
#pragma unroll
    for (int t = 0; t < 8; ++t)
      acc[t] = wmma_bf16(ap, load_frag_bt(vtbase + (size_t)(t * 16) * S_ + kk, S_),
                         acc[t]);
    __syncthreads();
  }

  float inv[8];
#pragma unroll
  for (int j = 0; j < 8; ++j) inv[j] = 1.0f / l_i[j];
#pragma unroll
  for (int t = 0; t < 8; ++t) {
#pragma unroll
    for (int j = 0; j < 8; ++j) {
      const float val = acc[t][j] * inv[j];
      const int s_ = qt * 16 + j + 8 * hi;
      const int hcol = head * DH_ + t * 16 + n;
      const size_t idx = ((size_t)b * S_ + s_) * H_ + hcol;
      ctx_out[idx] = val;
      ctx_bf[idx] = f2bf(val);
    }
  }
}

// ---------------- kernel 5: output projection (block: 128x64, 8 waves) ----------------
__global__ __launch_bounds__(256) void oproj_kernel(const unsigned short* __restrict__ ctx_bf,
                                                    const unsigned short* __restrict__ owT,
                                                    float* __restrict__ out) {
  __shared__ unsigned short bstage[2 * BROWS * BSTRIDE];
  const int wave = threadIdx.x >> 5;
  const int lane = threadIdx.x & 31;
  const int n = lane & 15, hi = lane >> 4;
  const int m0 = blockIdx.x * 128 + wave * 16;
  const int n0 = blockIdx.y * 64;

  v8f c[4];
#pragma unroll
  for (int t = 0; t < 4; ++t) c[t] = vzero8();

  gemm_core_128x64(ctx_bf + (size_t)m0 * H_, owT + (size_t)n0 * H_, bstage, c);

#pragma unroll
  for (int t = 0; t < 4; ++t) {
#pragma unroll
    for (int j = 0; j < 8; ++j) {
      const int r = m0 + j + 8 * hi;
      out[(size_t)r * H_ + n0 + 16 * t + n] = c[t][j];
    }
  }
}

// ---------------- host launcher ----------------
extern "C" void kernel_launch(void* const* d_in, const int* in_sizes, int n_in,
                              void* d_out, int out_size, void* d_ws, size_t ws_size,
                              hipStream_t stream) {
  const float* x     = (const float*)d_in[0];
  const float* mask  = (const float*)d_in[1];
  const float* qkvw  = (const float*)d_in[2];
  const float* qkvb  = (const float*)d_in[3];
  const float* ow    = (const float*)d_in[4];
  const float* nw    = (const float*)d_in[5];
  const float* nb    = (const float*)d_in[6];

  const size_t NEL = (size_t)ROWS_ * H_;         // 8,388,608 per tensor
  float* out_p  = (float*)d_out;                 // [B,S,H]
  float* k_out  = out_p + NEL;                   // [B,HEADS,S,dh]
  float* v_out  = out_p + 2 * NEL;               // [B,HEADS,S,dh]
  float* ctx_o  = out_p + 3 * NEL;               // [B,S,H]
  float* xn_o   = out_p + 4 * NEL;               // [B,S,H]

  unsigned short* ws = (unsigned short*)d_ws;
  unsigned short* xn_bf  = ws;                                     // ROWS*H
  unsigned short* qkvwT  = xn_bf + NEL;                            // 3H*H
  unsigned short* owT    = qkvwT + (size_t)H3_ * H_;               // H*H
  unsigned short* q_bf   = owT + (size_t)H_ * H_;                  // B*HEADS*S*dh
  unsigned short* k_bf   = q_bf + NEL;
  unsigned short* vT_bf  = k_bf + NEL;
  unsigned short* ctx_bf = vT_bf + NEL;

  // 1) fused layernorm (fp32 out + bf16 staging)
  ln_kernel<<<ROWS_, 256, 0, stream>>>(x, nw, nb, xn_o, xn_bf);

  // 2) weight bf16 transpose (qkvw^T, ow^T)
  {
    const size_t tot = (size_t)H3_ * H_ + (size_t)H_ * H_;
    const int blocks = (int)((tot + 255) / 256);
    wconv_kernel<<<blocks, 256, 0, stream>>>(qkvw, ow, qkvwT, owT);
  }

  // 3) QKV projection: [4096x2048] x [2048x6144] + bias, scatter q/k/v
  qkv_gemm_kernel<<<dim3(ROWS_ / 128, H3_ / 64), 256, 0, stream>>>(
      xn_bf, qkvwT, qkvb, k_out, v_out, q_bf, k_bf, vT_bf);

  // 4) causal flash attention, one wave per 16-query tile
  attn_kernel<<<B_ * HEADS_ * (S_ / 16), 32, 0, stream>>>(
      mask, q_bf, k_bf, vT_bf, ctx_o, ctx_bf);

  // 5) output projection: [4096x2048] x [2048x2048]
  oproj_kernel<<<dim3(ROWS_ / 128, H_ / 64), 256, 0, stream>>>(ctx_bf, owT, out_p);
}
